// VectorQuantizer_3874060501064
// MI455X (gfx1250) — matrix operand order repaired and checked by reading, not compiled, not run on hardware
//
#include <hip/hip_runtime.h>

// ---------------------------------------------------------------------------
// VQ-VAE vector quantizer for MI455X (gfx1250, wave32, WMMA)
//   x:     [32, 64, 64, 64] f32  (B, C=D, H, W)
//   embed: [1024, 64] f32
//   out:   quantized [32,64,64,64] f32  ++  loss scalar
// Distances via split-fp16 WMMA (xh*eh + xh*el + xl*eh, f32 accum, ~2^-22 rel).
// Codebook pre-scaled by -2048 (exact power-of-2; keeps e_lo out of fp16
// denormals); epilogue rescales with d = fma(acc, 2^-10, ||e||^2).
// M=64 rows/wave to amortize LDS B-fragment traffic over 24 WMMAs/tile.
// ---------------------------------------------------------------------------

#define D_EMBED 64
#define N_CODES 1024
#define HW      4096                 // 64*64
#define IMG     (D_EMBED * HW)       // floats per image = 262144
#define N_ROWS  131072               // 32 * 4096
#define ROWS_PER_WG   512
#define ROWS_PER_WAVE 64
#define MSUB    4                    // 16-row M-subtiles per wave
#define NWG     (N_ROWS / ROWS_PER_WG)   // 256
#define Q_ELEMS 8388608                  // 32*64*64*64

typedef __attribute__((ext_vector_type(16))) _Float16 v16h;
typedef __attribute__((ext_vector_type(8)))  float    v8f;

union Frag32B { v16h v; uint4 q[2]; };

__device__ __forceinline__ unsigned long long shflx64(unsigned long long v, int m) {
  unsigned int lo = (unsigned int)v, hi = (unsigned int)(v >> 32);
  lo = __shfl_xor(lo, m, 32);
  hi = __shfl_xor(hi, m, 32);
  return ((unsigned long long)hi << 32) | lo;
}

// ---------------------------------------------------------------------------
// Kernel 1: split (-2048 * codebook) into fp16 hi/lo terms + per-code ||e||^2
// ---------------------------------------------------------------------------
__global__ void vq_prep(const float* __restrict__ embed,
                        unsigned short* __restrict__ e_hi,
                        unsigned short* __restrict__ e_lo,
                        float* __restrict__ norms) {
  int k = blockIdx.x * blockDim.x + threadIdx.x;
  if (k >= N_CODES) return;
  const float* row = embed + k * D_EMBED;
  float nrm = 0.f;
#pragma unroll 8
  for (int c = 0; c < D_EMBED; ++c) {
    float v  = row[c];
    float v2 = -2048.0f * v;                    // exact pow2; folds -2*2^10
    _Float16 h = (_Float16)v2;                  // RNE via hw cvt
    _Float16 l = (_Float16)(v2 - (float)h);
    e_hi[k * D_EMBED + c] = __builtin_bit_cast(unsigned short, h);
    e_lo[k * D_EMBED + c] = __builtin_bit_cast(unsigned short, l);
    nrm = fmaf(v, v, nrm);
  }
  norms[k] = nrm;
}

// ---------------------------------------------------------------------------
// Kernel 2: distances via split-fp16 WMMA, argmin, gather, transpose, loss
// ---------------------------------------------------------------------------
__global__ __launch_bounds__(256) void vq_main(
    const float* __restrict__ x, const float* __restrict__ embed,
    const unsigned short* __restrict__ e_hi, const unsigned short* __restrict__ e_lo,
    const float* __restrict__ norms, float* __restrict__ qout,
    float* __restrict__ partials) {
  extern __shared__ char smem[];
  unsigned short* cb_hi = (unsigned short*)smem;              // 512 codes * 64 = 64 KB
  unsigned short* cb_lo = (unsigned short*)(smem + 65536);    // 64 KB
  float*        norm_s  = (float*)(smem + 131072);            // 4 KB (all 1024)
  unsigned int* idx_s   = (unsigned int*)(smem + 135168);     // 2 KB (512 rows)
  float*        red_s   = (float*)(smem + 137216);            // 8 floats

  const int tid  = threadIdx.x;
  const int lane = tid & 31;
  const int wave = tid >> 5;
  const int nloc = lane & 15;       // code-in-tile (B/N) / row-in-tile (A/M)
  const int hi16 = lane >> 4;

  for (int i = tid; i < N_CODES; i += 256) norm_s[i] = norms[i];

  const int n0    = blockIdx.x * ROWS_PER_WG;
  const int wrow0 = n0 + wave * ROWS_PER_WAVE;

  // ---- Load + split this wave's 64 x-rows into A fragments (registers) ----
  // 16-bit A 16x32 layout used here (contiguous k, not ISA table order): lane l -> M = l%16; K = 16*(l/16) .. +15
  v16h a_hi[MSUB][2], a_lo[MSUB][2];
  const int koff = hi16 * 8;
#pragma unroll
  for (int m = 0; m < MSUB; ++m) {
    int row = wrow0 + m * 16 + nloc;
    const float* xb = x + (size_t)(row >> 12) * IMG + (row & (HW - 1));
#pragma unroll
    for (int s = 0; s < 2; ++s) {
      v16h hv, lv;
#pragma unroll
      for (int e = 0; e < 16; ++e) {
        int c = s * 32 + hi16 * 16 + e;
        float f = xb[(size_t)c * HW];
        _Float16 hb = (_Float16)f;              // hw RNE cvt
        hv[e] = hb;
        lv[e] = (_Float16)(f - (float)hb);
      }
      a_hi[m][s] = hv;
      a_lo[m][s] = lv;
    }
  }

  float        bestd[MSUB][8];
  unsigned int besti[MSUB][8];
#pragma unroll
  for (int m = 0; m < MSUB; ++m)
#pragma unroll
    for (int j = 0; j < 8; ++j) { bestd[m][j] = __uint_as_float(0x7F800000u); besti[m][j] = 0u; }

  // Prefetch second codebook half (global_prefetch_b8) while first is used
  {
    const char* p0 = (const char*)(e_hi + 512 * D_EMBED);
    const char* p1 = (const char*)(e_lo + 512 * D_EMBED);
    for (int i = tid; i < 1024; i += 256) {       // 64 KB / 64 B lines, x2 arrays
      __builtin_prefetch(p0 + i * 64, 0, 1);
      __builtin_prefetch(p1 + i * 64, 0, 1);
    }
  }

#pragma unroll 1
  for (int half = 0; half < 2; ++half) {
    __syncthreads();  // protect LDS reuse between halves (and norm staging)
    {   // stage 512 codes (hi+lo) into LDS: 2 * 64 KB, coalesced b128
      const uint4* gh = (const uint4*)(e_hi + half * 512 * D_EMBED);
      const uint4* gl = (const uint4*)(e_lo + half * 512 * D_EMBED);
      uint4* sh = (uint4*)cb_hi;
      uint4* sl = (uint4*)cb_lo;
#pragma unroll 1
      for (int i = tid; i < 4096; i += 256) { sh[i] = gh[i]; sl[i] = gl[i]; }
    }
    __syncthreads();

    // B 32x16 fp16 layout: lane l -> N = l%16, K = (l<16 ? 0..15 : 16..31)
    // => 32 contiguous bytes per lane from row-major LDS codebook.
    // Double-buffered: fetch tile tl+1 while tile tl runs through the MMAs.
    Frag32B bh[2][2], bl[2][2];
    float   nbuf[2];
    auto loadB = [&](int bi, int tlv) {
      const int byte0 = (tlv * 16 + nloc) * 128 + hi16 * 32;
#pragma unroll
      for (int s = 0; s < 2; ++s) {
        const uint4* ph = (const uint4*)((const char*)cb_hi + byte0 + s * 64);
        const uint4* pl = (const uint4*)((const char*)cb_lo + byte0 + s * 64);
        bh[bi][s].q[0] = ph[0]; bh[bi][s].q[1] = ph[1];
        bl[bi][s].q[0] = pl[0]; bl[bi][s].q[1] = pl[1];
      }
      nbuf[bi] = norm_s[(half * 32 + tlv) * 16 + nloc];
    };
    loadB(0, 0);

#pragma unroll 2
    for (int tl = 0; tl < 32; ++tl) {
      const int cur = tl & 1;
      if (tl < 31) loadB(cur ^ 1, tl + 1);

      const int t = half * 32 + tl;
      const float nbb = nbuf[cur];
      const unsigned int code = (unsigned int)(t * 16 + nloc);
#pragma unroll
      for (int m = 0; m < MSUB; ++m) {
        v8f acc = {0.f, 0.f, 0.f, 0.f, 0.f, 0.f, 0.f, 0.f};
#pragma unroll
        for (int s = 0; s < 2; ++s) {   // (-2048e) split: xl*eh + xh*el + xh*eh
          acc = __builtin_amdgcn_wmma_f32_16x16x32_f16(
              false, a_lo[m][s], false, bh[cur][s].v, (short)0, acc, false, false);
          acc = __builtin_amdgcn_wmma_f32_16x16x32_f16(
              false, a_hi[m][s], false, bl[cur][s].v, (short)0, acc, false, false);
          acc = __builtin_amdgcn_wmma_f32_16x16x32_f16(
              false, a_hi[m][s], false, bh[cur][s].v, (short)0, acc, false, false);
        }
#pragma unroll
        for (int j = 0; j < 8; ++j) {
          // d = ||e||^2 - 2 x.e  (acc carries the extra 2^10; exact rescale)
          float d = fmaf(acc[j], 9.765625e-4f, nbb);
          bool lt = d < bestd[m][j];              // strict <: first-occurrence
          bestd[m][j] = lt ? d : bestd[m][j];
          besti[m][j] = lt ? code : besti[m][j];
        }
      }
    }
  }

  // ---- argmin reduce across the 16 lanes sharing each row; publish index ----
#pragma unroll
  for (int m = 0; m < MSUB; ++m) {
#pragma unroll
    for (int j = 0; j < 8; ++j) {
      unsigned int u   = __float_as_uint(bestd[m][j]);
      unsigned int key = (u & 0x80000000u) ? ~u : (u | 0x80000000u);
      unsigned long long k = ((unsigned long long)key << 32) | besti[m][j];
#pragma unroll
      for (int off = 1; off < 16; off <<= 1) {
        unsigned long long o = shflx64(k, off);
        if (o < k) k = o;
      }
      if (nloc == 0) {   // lanes 0 (rows M=j) and 16 (rows M=j+8)
        int rloc = wave * ROWS_PER_WAVE + m * 16 + j + hi16 * 8;
        idx_s[rloc] = (unsigned int)(k & 0xFFFFFFFFu);
      }
    }
  }
  __syncthreads();

  // ---- gather winning codes (b128), transposed write-back, loss partial ----
  float ls = 0.f;
#pragma unroll
  for (int r = 0; r < 2; ++r) {
    const int rloc = tid + r * 256;
    const int row  = n0 + rloc;
    const unsigned int ci = idx_s[rloc];
    const float4* eq = (const float4*)(embed + (size_t)ci * D_EMBED);
    const size_t obase = (size_t)(row >> 12) * IMG + (row & (HW - 1));
#pragma unroll 4
    for (int c4 = 0; c4 < 16; ++c4) {
      float4 q = eq[c4];
      const size_t o = obase + (size_t)(c4 * 4) * HW;
      float x0 = x[o];             float x1 = x[o + HW];
      float x2 = x[o + 2 * HW];    float x3 = x[o + 3 * HW];
      qout[o]          = q.x;      qout[o + HW]     = q.y;
      qout[o + 2 * HW] = q.z;      qout[o + 3 * HW] = q.w;
      float d0 = q.x - x0, d1 = q.y - x1, d2 = q.z - x2, d3 = q.w - x3;
      ls = fmaf(d0, d0, ls); ls = fmaf(d1, d1, ls);
      ls = fmaf(d2, d2, ls); ls = fmaf(d3, d3, ls);
    }
  }
#pragma unroll
  for (int off = 16; off > 0; off >>= 1) ls += __shfl_xor(ls, off, 32);
  if (lane == 0) red_s[wave] = ls;
  __syncthreads();
  if (tid == 0) {
    float s = 0.f;
#pragma unroll
    for (int w = 0; w < 8; ++w) s += red_s[w];
    partials[blockIdx.x] = s;
  }
}

// ---------------------------------------------------------------------------
// Kernel 3: deterministic fixed-order final loss reduction
// ---------------------------------------------------------------------------
__global__ void vq_loss(const float* __restrict__ partials,
                        float* __restrict__ loss_out) {
  if (threadIdx.x == 0 && blockIdx.x == 0) {
    float s = 0.f;
    for (int i = 0; i < NWG; ++i) s += partials[i];
    // q_latent_loss + 0.25 * e_latent_loss, both == mean((q-x)^2)
    loss_out[0] = (s / (float)Q_ELEMS) * 1.25f;
  }
}

// ---------------------------------------------------------------------------
extern "C" void kernel_launch(void* const* d_in, const int* in_sizes, int n_in,
                              void* d_out, int out_size, void* d_ws, size_t ws_size,
                              hipStream_t stream) {
  (void)in_sizes; (void)n_in; (void)out_size; (void)ws_size;
  const float* x     = (const float*)d_in[0];
  const float* embed = (const float*)d_in[1];
  float* qout = (float*)d_out;
  float* loss = qout + Q_ELEMS;

  char* ws = (char*)d_ws;                       // ~268 KB used
  unsigned short* e_hi    = (unsigned short*)(ws);
  unsigned short* e_lo    = (unsigned short*)(ws + 131072);
  float*          norms   = (float*)(ws + 262144);
  float*          partial = (float*)(ws + 266240);

  vq_prep<<<(N_CODES + 255) / 256, 256, 0, stream>>>(embed, e_hi, e_lo, norms);
  vq_main<<<NWG, 256, 137248, stream>>>(x, embed, e_hi, e_lo, norms, qout, partial);
  vq_loss<<<1, 32, 0, stream>>>(partial, loss);
}
